// MultiHeadDiffAttention_86354612453685
// MI455X (gfx1250) — compile-verified, matrix operand order checked by reading, not run on hardware
//
#include <hip/hip_runtime.h>
#include <hip/hip_bf16.h>

// ---------------------------------------------------------------------------
// Differential Multi-Head Attention for MI455X (gfx1250), bf16 WMMA pipeline
// with async global->LDS tile movement (GLOBAL_LOAD_ASYNC_TO_LDS_B128).
// B=2, T=2048, C=1024, H=16, HD=64, 2HD=128.
// ---------------------------------------------------------------------------

typedef __attribute__((ext_vector_type(16))) __bf16 v16bf;
typedef __attribute__((ext_vector_type(8)))  float  v8f;

#define T_SEQ   2048
#define C_DIM   1024
#define H_NUM   16
#define HD_DIM  64
#define D2_DIM  128
#define BT_M    4096            // B*T
#define LAMBDA_INIT 0.25550907f // 0.7 - 0.6*exp(-0.3)
#define LN_EPS  1e-5f
#define QK_SCALE 0.125f         // 1/sqrt(64)

// ---- workspace layout (bytes) ----
#define OFF_XBF  ((size_t)0)          // 4096*1024 bf16      =  8 MiB
#define OFF_WQ1  ((size_t)8388608)    // 1024*1024 bf16      =  2 MiB
#define OFF_WQ2  ((size_t)10485760)
#define OFF_WK1  ((size_t)12582912)
#define OFF_WK2  ((size_t)14680064)
#define OFF_WV   ((size_t)16777216)   // 2048*1024 bf16      =  4 MiB
#define OFF_WC   ((size_t)20971520)   // 1024*2048 bf16      =  4 MiB
#define OFF_Q1   ((size_t)25165824)   // [B,H,T,64] bf16     =  8 MiB
#define OFF_Q2   ((size_t)33554432)
#define OFF_K1   ((size_t)41943040)
#define OFF_K2   ((size_t)50331648)
#define OFF_VT   ((size_t)58720256)   // [B,H,128,T] bf16    = 16 MiB
#define OFF_Y    ((size_t)75497472)   // [B,T,2048] bf16     = 16 MiB
#define OFF_LAM  ((size_t)92274688)   // 16 floats

__device__ __forceinline__ __bf16 f2bf(float f) {
    union { float f; unsigned u; } c; c.f = f;
    unsigned r = (c.u + 0x7FFFu + ((c.u >> 16) & 1u)) >> 16;
    unsigned short s = (unsigned short)r;
    __bf16 b;
    __builtin_memcpy(&b, &s, 2);
    return b;
}

// LDS byte offset of a generic pointer to a __shared__ object:
// generic = (shared_aperture_hi << 32) | lds_offset  ->  truncate to 32 bits.
__device__ __forceinline__ unsigned lds_off(const void* p) {
    return (unsigned)(uintptr_t)p;
}

// Async global->LDS 16B copy (per lane), GVS addressing:
//   mem = SGPR_base64 + VGPR_byteoff32 ; LDS dest = VDST vgpr (byte address).
// Tracked by ASYNCcnt.
__device__ __forceinline__ void async_ld128(unsigned ldsAddr, const void* base,
                                            unsigned byteOff) {
    asm volatile("global_load_async_to_lds_b128 %0, %1, %2"
                 :: "v"(ldsAddr), "v"(byteOff), "s"(base)
                 : "memory");
}
__device__ __forceinline__ void wait_async0() {
    asm volatile("s_wait_asynccnt 0x0" ::: "memory");
}

// Shared fragment loader: for C = A * W^T with A[M,K], W[N,K] row-major,
// A and B fragments use the same (contiguous) k order in this code:
// lane l reads 16 contiguous K elements of row (base + l%16), K-half (l/16).
__device__ __forceinline__ v16bf frag_ld(const __bf16* base, int row0, int stride,
                                         int k0, int lane) {
    const __bf16* p = base + (size_t)(row0 + (lane & 15)) * stride
                           + k0 + ((lane >> 4) << 4);
    return *(const v16bf*)p;
}

__device__ __forceinline__ v8f wmma_bf16(v16bf a, v16bf b, v8f c) {
    return __builtin_amdgcn_wmma_f32_16x16x32_bf16(false, a, false, b,
                                                   (short)0, c, false, false);
}

// ---------------------------------------------------------------------------
// fp32 -> bf16 cast
// ---------------------------------------------------------------------------
__global__ __launch_bounds__(256) void cast_f32_bf16(const float* __restrict__ src,
                                                     __bf16* __restrict__ dst, int n) {
    int i = blockIdx.x * 256 + threadIdx.x;
    if (i < n) dst[i] = f2bf(src[i]);
}

// ---------------------------------------------------------------------------
// lambda_h = exp(lq1.lk1) - exp(lq2.lk2) + LAMBDA_INIT
// ---------------------------------------------------------------------------
__global__ void lambda_kernel(const float* __restrict__ lq1, const float* __restrict__ lk1,
                              const float* __restrict__ lq2, const float* __restrict__ lk2,
                              float* __restrict__ lam) {
    int h = threadIdx.x;
    if (h < H_NUM) {
        float s1 = 0.f, s2 = 0.f;
        for (int j = 0; j < HD_DIM; j++) {
            s1 += lq1[h * HD_DIM + j] * lk1[h * HD_DIM + j];
            s2 += lq2[h * HD_DIM + j] * lk2[h * HD_DIM + j];
        }
        lam[h] = __expf(s1) - __expf(s2) + LAMBDA_INIT;
    }
}

// ---------------------------------------------------------------------------
// GEMM: C[M,N] = A[M,K] (bf16, row-major) @ W[N,K]^T (bf16, row-major)
// 128x128 tile per block, BK=32, 8 waves, fp32 accum via v_wmma.
// Tile fills use async global->LDS.
// mode 0: fp32 row-major out
// mode 1: bf16, q/k head reshuffle -> [B,H,T,64]
// mode 2: bf16, V transposed       -> [B,H,128,T]
// ---------------------------------------------------------------------------
__global__ __launch_bounds__(256) void gemm_bf16(const __bf16* __restrict__ A,
                                                 const __bf16* __restrict__ W,
                                                 void* __restrict__ dst,
                                                 int M, int N, int K, int mode) {
    __shared__ __align__(32) __bf16 As[128 * 32];
    __shared__ __align__(32) __bf16 Ws[128 * 32];

    const int tid  = threadIdx.x;
    const int lane = tid & 31;
    const int wv   = tid >> 5;   // 0..7
    const int wm   = wv >> 2;    // 0..1  (64 rows)
    const int wn   = wv & 3;     // 0..3  (32 cols)
    const int mTile = blockIdx.y * 128;
    const int nTile = blockIdx.x * 128;
    const unsigned ldsA0 = lds_off(As);
    const unsigned ldsW0 = lds_off(Ws);

    v8f acc[4][2] = {};

    for (int k0 = 0; k0 < K; k0 += 32) {
#pragma unroll
        for (int i = 0; i < 2; i++) {
            int c   = tid + 256 * i;      // 0..511 chunk slots of 8 bf16
            int row = c >> 2, cb = c & 3;
            unsigned offA = (unsigned)((mTile + row) * K + k0 + cb * 8) * 2u;
            unsigned offW = (unsigned)((nTile + row) * K + k0 + cb * 8) * 2u;
            async_ld128(ldsA0 + (unsigned)c * 16u, A, offA);
            async_ld128(ldsW0 + (unsigned)c * 16u, W, offW);
        }
        if (k0 + 32 < K) { // prefetch next K-slab (global_prefetch_b8)
            __builtin_prefetch(A + (size_t)(mTile + (tid >> 1)) * K + k0 + 32, 0, 1);
            __builtin_prefetch(W + (size_t)(nTile + (tid >> 1)) * K + k0 + 32, 0, 1);
        }
        wait_async0();
        __syncthreads();

        v16bf aF[4], bF[2];
#pragma unroll
        for (int mi = 0; mi < 4; mi++) aF[mi] = frag_ld(As, wm * 64 + mi * 16, 32, 0, lane);
#pragma unroll
        for (int ni = 0; ni < 2; ni++) bF[ni] = frag_ld(Ws, wn * 32 + ni * 16, 32, 0, lane);
#pragma unroll
        for (int mi = 0; mi < 4; mi++)
#pragma unroll
            for (int ni = 0; ni < 2; ni++)
                acc[mi][ni] = wmma_bf16(aF[mi], bF[ni], acc[mi][ni]);
        __syncthreads();
    }

    // epilogue: lane l, vgpr e -> M = e + 8*(l>>4), N = l&15
#pragma unroll
    for (int mi = 0; mi < 4; mi++)
#pragma unroll
        for (int ni = 0; ni < 2; ni++)
#pragma unroll
            for (int e = 0; e < 8; e++) {
                int row = mTile + wm * 64 + mi * 16 + e + ((lane >> 4) << 3);
                int col = nTile + wn * 32 + ni * 16 + (lane & 15);
                float v = acc[mi][ni][e];
                if (mode == 0) {
                    ((float*)dst)[(size_t)row * N + col] = v;
                } else if (mode == 1) {
                    int b = row >> 11, t = row & (T_SEQ - 1);
                    int h = col >> 6,  d = col & 63;
                    ((__bf16*)dst)[((size_t)((b * H_NUM + h) * T_SEQ + t)) * 64 + d] = f2bf(v);
                } else {
                    int b = row >> 11, t = row & (T_SEQ - 1);
                    int h = col >> 7,  d = col & 127;
                    ((__bf16*)dst)[((size_t)((b * H_NUM + h) * 128 + d)) * T_SEQ + t] = f2bf(v);
                }
            }
}

// ---------------------------------------------------------------------------
// S tile helper: sS[64x128] = sQ[64x64] @ sK[128x64]^T  (4 waves, wave w owns
// key columns 32w..32w+31)
// ---------------------------------------------------------------------------
__device__ __forceinline__ void compute_S(const __bf16* sQ, const __bf16* sK,
                                          float* sS, int lane, int wv) {
    v8f acc[4][2] = {};
#pragma unroll
    for (int ks = 0; ks < 2; ks++) {
        int k0 = ks * 32;
        v16bf aF[4], bF[2];
#pragma unroll
        for (int mi = 0; mi < 4; mi++) aF[mi] = frag_ld(sQ, mi * 16, 64, k0, lane);
#pragma unroll
        for (int ni = 0; ni < 2; ni++) bF[ni] = frag_ld(sK, wv * 32 + ni * 16, 64, k0, lane);
#pragma unroll
        for (int mi = 0; mi < 4; mi++)
#pragma unroll
            for (int ni = 0; ni < 2; ni++)
                acc[mi][ni] = wmma_bf16(aF[mi], bF[ni], acc[mi][ni]);
    }
#pragma unroll
    for (int mi = 0; mi < 4; mi++)
#pragma unroll
        for (int ni = 0; ni < 2; ni++)
#pragma unroll
            for (int e = 0; e < 8; e++) {
                int row = mi * 16 + e + ((lane >> 4) << 3);
                int col = wv * 32 + ni * 16 + (lane & 15);
                sS[row * 128 + col] = acc[mi][ni][e];
            }
}

// ---------------------------------------------------------------------------
// Fused differential attention per (b, h, 64-row query tile).
// Two-pass safe softmax (phase A: stats, phase B: P = p1 - lam*p2, O += P@V),
// then 128-wide sub-LayerNorm * (1 - lambda_init) -> Y[B,T,2048] bf16.
// 128 threads = 4 waves. Tile fills use async global->LDS.
// ---------------------------------------------------------------------------
__global__ __launch_bounds__(128) void diffattn_kernel(
    const __bf16* __restrict__ Q1g, const __bf16* __restrict__ Q2g,
    const __bf16* __restrict__ K1g, const __bf16* __restrict__ K2g,
    const __bf16* __restrict__ Vtg, __bf16* __restrict__ Yg,
    const float* __restrict__ lamArr) {

    __shared__ __align__(32) char smem[115712];
    __bf16* sQ1 = (__bf16*)(smem + 0);         //  8 KiB  64x64
    __bf16* sQ2 = (__bf16*)(smem + 8192);      //  8 KiB  64x64
    __bf16* sK  = (__bf16*)(smem + 16384);     // 16 KiB  128x64 (K1 then K2)
    float*  sS1 = (float*)(smem + 32768);      // 32 KiB  64x128   } aliased
    __bf16* sV  = (__bf16*)(smem + 32768);     // 32 KiB  128x128  } with sS1
    float*  sOb = (float*)(smem + 32768);      // 32 KiB  64x128   } region
    float*  sS2 = (float*)(smem + 65536);      // 32 KiB  64x128
    __bf16* sP  = (__bf16*)(smem + 98304);     // 16 KiB  64x128
    float*  sM1 = (float*)(smem + 114688);
    float*  sL1 = sM1 + 64;
    float*  sM2 = sM1 + 128;
    float*  sL2 = sM1 + 192;

    const int tid  = threadIdx.x;
    const int lane = tid & 31;
    const int wv   = tid >> 5;                 // 0..3
    const int qt = blockIdx.x & 31;
    const int h  = (blockIdx.x >> 5) & 15;
    const int b  = blockIdx.x >> 9;
    const int qBase = qt * 64;

    const size_t headQK = ((size_t)(b * H_NUM + h)) * T_SEQ * 64;
    const __bf16* Q1p = Q1g + headQK + (size_t)qBase * 64;
    const __bf16* Q2p = Q2g + headQK + (size_t)qBase * 64;
    const __bf16* K1p = K1g + headQK;
    const __bf16* K2p = K2g + headQK;
    const __bf16* Vtp = Vtg + ((size_t)(b * H_NUM + h)) * 128 * T_SEQ;
    const float lam = lamArr[h];

    const unsigned ldsQ1 = lds_off(sQ1);
    const unsigned ldsQ2 = lds_off(sQ2);
    const unsigned ldsK  = lds_off(sK);
    const unsigned ldsV  = lds_off(sV);

    // async load Q1/Q2 tiles (64x64 bf16 each; tiles are contiguous 8 KiB)
#pragma unroll
    for (int i = 0; i < 4; i++) {
        unsigned c16 = (unsigned)(tid + 128 * i) * 16u;   // 0..8176
        async_ld128(ldsQ1 + c16, Q1p, c16);
        async_ld128(ldsQ2 + c16, Q2p, c16);
    }

    const int ktEnd = (qBase + 63) >> 7;       // key tiles of 128
    float rm1 = -1e30f, rl1 = 0.f, rm2 = -1e30f, rl2 = 0.f;

    // -------- Phase A: softmax statistics --------
    for (int kt = 0; kt <= ktEnd; kt++) {
        const int kBase = kt * 128;
        const unsigned kOffB = (unsigned)kBase * 128u;    // kBase rows * 64 * 2B
        __syncthreads();
#pragma unroll
        for (int i = 0; i < 8; i++) {          // K1 tile 128x64 (contig 16 KiB)
            unsigned c16 = (unsigned)(tid + 128 * i) * 16u;
            async_ld128(ldsK + c16, K1p, kOffB + c16);
        }
        wait_async0();
        __syncthreads();
        compute_S(sQ1, sK, sS1, lane, wv);
        __syncthreads();
#pragma unroll
        for (int i = 0; i < 8; i++) {          // K2 tile
            unsigned c16 = (unsigned)(tid + 128 * i) * 16u;
            async_ld128(ldsK + c16, K2p, kOffB + c16);
        }
        wait_async0();
        __syncthreads();
        compute_S(sQ2, sK, sS2, lane, wv);
        __syncthreads();

        if (tid < 64) {
            const int qg = qBase + tid;
            float mN = rm1;
            for (int c = 0; c < 128; c++)
                if (kBase + c <= qg) mN = fmaxf(mN, sS1[tid * 128 + c] * QK_SCALE);
            float sum = 0.f;
            for (int c = 0; c < 128; c++)
                if (kBase + c <= qg) sum += __expf(sS1[tid * 128 + c] * QK_SCALE - mN);
            rl1 = rl1 * __expf(rm1 - mN) + sum;
            rm1 = mN;

            mN = rm2;
            for (int c = 0; c < 128; c++)
                if (kBase + c <= qg) mN = fmaxf(mN, sS2[tid * 128 + c] * QK_SCALE);
            sum = 0.f;
            for (int c = 0; c < 128; c++)
                if (kBase + c <= qg) sum += __expf(sS2[tid * 128 + c] * QK_SCALE - mN);
            rl2 = rl2 * __expf(rm2 - mN) + sum;
            rm2 = mN;
        }
    }
    if (tid < 64) { sM1[tid] = rm1; sL1[tid] = rl1; sM2[tid] = rm2; sL2[tid] = rl2; }
    __syncthreads();

    // -------- Phase B: O = (softmax1 - lam*softmax2) @ V --------
    v8f oAcc[4][2] = {};
    for (int kt = 0; kt <= ktEnd; kt++) {
        const int kBase = kt * 128;
        const unsigned kOffB = (unsigned)kBase * 128u;
        __syncthreads();
#pragma unroll
        for (int i = 0; i < 8; i++) {
            unsigned c16 = (unsigned)(tid + 128 * i) * 16u;
            async_ld128(ldsK + c16, K1p, kOffB + c16);
        }
        wait_async0();
        __syncthreads();
        compute_S(sQ1, sK, sS1, lane, wv);
        __syncthreads();
#pragma unroll
        for (int i = 0; i < 8; i++) {
            unsigned c16 = (unsigned)(tid + 128 * i) * 16u;
            async_ld128(ldsK + c16, K2p, kOffB + c16);
        }
        wait_async0();
        __syncthreads();
        compute_S(sQ2, sK, sS2, lane, wv);
        __syncthreads();

        // P tile: column tid, all 64 rows
#pragma unroll 4
        for (int i = 0; i < 64; i++) {
            float p = 0.f;
            if (kBase + tid <= qBase + i) {
                float p1 = __expf(sS1[i * 128 + tid] * QK_SCALE - sM1[i]) / sL1[i];
                float p2 = __expf(sS2[i * 128 + tid] * QK_SCALE - sM2[i]) / sL2[i];
                p = p1 - lam * p2;
            }
            sP[i * 128 + tid] = f2bf(p);
        }
        __syncthreads();

        // V tile (transposed layout: [dim][key]) -> sV (overwrites sS1)
#pragma unroll
        for (int i = 0; i < 16; i++) {
            int c = tid + 128 * i;             // 0..2047
            int row = c >> 4, cb = c & 15;
            unsigned gOff = (unsigned)row * 4096u + ((unsigned)kBase << 1) + (unsigned)cb * 16u;
            async_ld128(ldsV + (unsigned)c * 16u, Vtp, gOff);
        }
        wait_async0();
        __syncthreads();

        // O += P[64x128] @ V[128x128]
#pragma unroll
        for (int ks = 0; ks < 4; ks++) {
            int k0 = ks * 32;
            v16bf aF[4], bF[2];
#pragma unroll
            for (int mi = 0; mi < 4; mi++) aF[mi] = frag_ld(sP, mi * 16, 128, k0, lane);
#pragma unroll
            for (int ni = 0; ni < 2; ni++) bF[ni] = frag_ld(sV, wv * 32 + ni * 16, 128, k0, lane);
#pragma unroll
            for (int mi = 0; mi < 4; mi++)
#pragma unroll
                for (int ni = 0; ni < 2; ni++)
                    oAcc[mi][ni] = wmma_bf16(aF[mi], bF[ni], oAcc[mi][ni]);
        }
    }

    __syncthreads();
#pragma unroll
    for (int mi = 0; mi < 4; mi++)
#pragma unroll
        for (int ni = 0; ni < 2; ni++)
#pragma unroll
            for (int e = 0; e < 8; e++) {
                int row = mi * 16 + e + ((lane >> 4) << 3);
                int col = wv * 32 + ni * 16 + (lane & 15);
                sOb[row * 128 + col] = oAcc[mi][ni][e];
            }
    __syncthreads();

    // sub-LayerNorm over 128 dims (no affine), * (1 - lambda_init)
    if (tid < 64) {
        float mu = 0.f;
        for (int c = 0; c < 128; c++) mu += sOb[tid * 128 + c];
        mu *= (1.f / 128.f);
        float var = 0.f;
        for (int c = 0; c < 128; c++) {
            float d = sOb[tid * 128 + c] - mu;
            var += d * d;
        }
        var *= (1.f / 128.f);
        float inv = rsqrtf(var + LN_EPS) * (1.0f - LAMBDA_INIT);
        __bf16* yrow = Yg + ((size_t)(b * T_SEQ + qBase + tid)) * 2048 + h * 128;
        for (int c = 0; c < 128; c++) yrow[c] = f2bf((sOb[tid * 128 + c] - mu) * inv);
    }
}

// ---------------------------------------------------------------------------
extern "C" void kernel_launch(void* const* d_in, const int* in_sizes, int n_in,
                              void* d_out, int out_size, void* d_ws, size_t ws_size,
                              hipStream_t stream) {
    (void)in_sizes; (void)n_in; (void)out_size; (void)ws_size;

    const float* x   = (const float*)d_in[0];
    const float* Wq1 = (const float*)d_in[1];
    const float* Wq2 = (const float*)d_in[2];
    const float* Wk1 = (const float*)d_in[3];
    const float* Wk2 = (const float*)d_in[4];
    const float* Wv  = (const float*)d_in[5];
    const float* Wc  = (const float*)d_in[6];
    const float* lq1 = (const float*)d_in[7];
    const float* lk1 = (const float*)d_in[8];
    const float* lq2 = (const float*)d_in[9];
    const float* lk2 = (const float*)d_in[10];

    char* ws = (char*)d_ws;
    __bf16* Xbf  = (__bf16*)(ws + OFF_XBF);
    __bf16* Wq1b = (__bf16*)(ws + OFF_WQ1);
    __bf16* Wq2b = (__bf16*)(ws + OFF_WQ2);
    __bf16* Wk1b = (__bf16*)(ws + OFF_WK1);
    __bf16* Wk2b = (__bf16*)(ws + OFF_WK2);
    __bf16* Wvb  = (__bf16*)(ws + OFF_WV);
    __bf16* Wcb  = (__bf16*)(ws + OFF_WC);
    __bf16* Q1   = (__bf16*)(ws + OFF_Q1);
    __bf16* Q2   = (__bf16*)(ws + OFF_Q2);
    __bf16* K1   = (__bf16*)(ws + OFF_K1);
    __bf16* K2   = (__bf16*)(ws + OFF_K2);
    __bf16* VT   = (__bf16*)(ws + OFF_VT);
    __bf16* Y    = (__bf16*)(ws + OFF_Y);
    float*  lam  = (float*)(ws + OFF_LAM);

    // fp32 -> bf16 casts
    cast_f32_bf16<<<(4194304 + 255) / 256, 256, 0, stream>>>(x, Xbf, 4194304);
    cast_f32_bf16<<<(1048576 + 255) / 256, 256, 0, stream>>>(Wq1, Wq1b, 1048576);
    cast_f32_bf16<<<(1048576 + 255) / 256, 256, 0, stream>>>(Wq2, Wq2b, 1048576);
    cast_f32_bf16<<<(1048576 + 255) / 256, 256, 0, stream>>>(Wk1, Wk1b, 1048576);
    cast_f32_bf16<<<(1048576 + 255) / 256, 256, 0, stream>>>(Wk2, Wk2b, 1048576);
    cast_f32_bf16<<<(2097152 + 255) / 256, 256, 0, stream>>>(Wv, Wvb, 2097152);
    cast_f32_bf16<<<(2097152 + 255) / 256, 256, 0, stream>>>(Wc, Wcb, 2097152);

    lambda_kernel<<<1, 16, 0, stream>>>(lq1, lk1, lq2, lk2, lam);

    // projections: [4096,1024] @ W^T
    gemm_bf16<<<dim3(8, 32), 256, 0, stream>>>(Xbf, Wq1b, Q1, BT_M, 1024, 1024, 1);
    gemm_bf16<<<dim3(8, 32), 256, 0, stream>>>(Xbf, Wq2b, Q2, BT_M, 1024, 1024, 1);
    gemm_bf16<<<dim3(8, 32), 256, 0, stream>>>(Xbf, Wk1b, K1, BT_M, 1024, 1024, 1);
    gemm_bf16<<<dim3(8, 32), 256, 0, stream>>>(Xbf, Wk2b, K2, BT_M, 1024, 1024, 1);
    gemm_bf16<<<dim3(16, 32), 256, 0, stream>>>(Xbf, Wvb, VT, BT_M, 2048, 1024, 2);

    // fused differential attention: B*H*(T/64) = 1024 blocks
    diffattn_kernel<<<1024, 128, 0, stream>>>(Q1, Q2, K1, K2, VT, Y, lam);

    // output projection: [4096,2048] @ Wc^T -> fp32 d_out
    gemm_bf16<<<dim3(8, 32), 256, 0, stream>>>(Y, Wcb, d_out, BT_M, 1024, 2048, 0);
}
